// Gemma2DecoderLayer_26757646254430
// MI455X (gfx1250) — matrix operand order repaired and checked by reading, not compiled, not run on hardware
//
#include <hip/hip_runtime.h>
#include <hip/hip_bf16.h>
#include <stdint.h>

// ---------------------------------------------------------------- types
typedef __bf16 bf16;
typedef __attribute__((ext_vector_type(16))) __bf16 v16bf;
typedef __attribute__((ext_vector_type(8)))  float  v8f;
typedef int v4i_gcc __attribute__((vector_size(16)));   // matches builtin's V4i

// ---------------------------------------------------------------- problem dims
constexpr int   HID   = 3584;
constexpr int   NH    = 16;
constexpr int   NKV   = 8;
constexpr int   HD    = 256;
constexpr int   FF    = 14336;
constexpr int   BB    = 2;
constexpr int   SEQ   = 2048;
constexpr int   MROWS = BB * SEQ;          // 4096 token rows
constexpr float EPSILON = 1e-6f;
constexpr float SCALE_Q = 0.0625f;         // 256^-0.5
constexpr float CAP     = 50.0f;
constexpr int   WIN     = 1024;

// ---------------------------------------------------------------- CDNA5 async global->LDS copy
#if defined(__HIP_DEVICE_COMPILE__) && __has_builtin(__builtin_amdgcn_global_load_async_to_lds_b128)
#define HAS_ASYNC_LDS 1
#else
#define HAS_ASYNC_LDS 0
#endif

__device__ __forceinline__ void cp16_g2l(const bf16* g, bf16* l) {
#if HAS_ASYNC_LDS
  __attribute__((address_space(1))) v4i_gcc* gp =
      (__attribute__((address_space(1))) v4i_gcc*)(
          (__attribute__((address_space(1))) void*)(void*)g);
  __attribute__((address_space(3))) v4i_gcc* lp =
      (__attribute__((address_space(3))) v4i_gcc*)(
          (__attribute__((address_space(3))) void*)(void*)l);
  __builtin_amdgcn_global_load_async_to_lds_b128(gp, lp, 0, 0);
#else
  *(uint4*)l = *(const uint4*)g;
#endif
}
__device__ __forceinline__ void wait_async_lds() {
#if HAS_ASYNC_LDS
#if __has_builtin(__builtin_amdgcn_s_wait_asynccnt)
  __builtin_amdgcn_s_wait_asynccnt(0);
#else
  asm volatile("s_wait_asynccnt 0x0" ::: "memory");
#endif
#endif
}

// ---------------------------------------------------------------- helpers
__device__ __forceinline__ bf16 f2bf(float f) {
  unsigned u = __builtin_bit_cast(unsigned, f);
  unsigned r = (u + 0x7FFFu + ((u >> 16) & 1u)) >> 16;   // RNE
  unsigned short h = (unsigned short)r;
  return __builtin_bit_cast(bf16, h);
}
__device__ __forceinline__ v16bf pack2(uint4 lo, uint4 hi) {
  union { uint4 u[2]; v16bf v; } t;
  t.u[0] = lo; t.u[1] = hi;
  return t.v;
}
__device__ __forceinline__ v8f wmma_bf16(v16bf a, v16bf b, v8f c) {
  return __builtin_amdgcn_wmma_f32_16x16x32_bf16(false, a, false, b, (short)0, c, false, false);
}
__device__ __forceinline__ float gelu_tanh(float x) {
  const float k0 = 0.7978845608028654f, k1 = 0.044715f;
  return 0.5f * x * (1.0f + tanhf(k0 * (x + k1 * x * x * x)));
}

// ---------------------------------------------------------------- fp32 -> bf16 convert
__global__ __launch_bounds__(256) void cvt_f32_bf16(const float* __restrict__ in,
                                                    bf16* __restrict__ out, long n) {
  long i = (long)blockIdx.x * 256 + threadIdx.x;
  long stride = (long)gridDim.x * 256;
  for (; i < n; i += stride) out[i] = f2bf(in[i]);
}

// ---------------------------------------------------------------- RMSNorm variants
__global__ __launch_bounds__(256) void rmsnorm_bf16_kernel(const float* __restrict__ x,
                                                           const float* __restrict__ wln,
                                                           bf16* __restrict__ out) {
  __shared__ float red[8];
  const size_t base = (size_t)blockIdx.x * HID;
  float ss = 0.f;
  for (int i = threadIdx.x; i < HID; i += 256) { float v = x[base + i]; ss += v * v; }
#pragma unroll
  for (int off = 16; off >= 1; off >>= 1) ss += __shfl_xor(ss, off, 32);
  if ((threadIdx.x & 31) == 0) red[threadIdx.x >> 5] = ss;
  __syncthreads();
  float tot = 0.f;
#pragma unroll
  for (int i = 0; i < 8; ++i) tot += red[i];
  const float inv = rsqrtf(tot / (float)HID + EPSILON);
  for (int i = threadIdx.x; i < HID; i += 256)
    out[base + i] = f2bf(x[base + i] * inv * (1.0f + wln[i]));
}

__global__ __launch_bounds__(256) void add_rmsnorm_f32(const float* __restrict__ x,
                                                       const float* __restrict__ resid,
                                                       const float* __restrict__ wln,
                                                       float* __restrict__ out) {
  __shared__ float red[8];
  const size_t base = (size_t)blockIdx.x * HID;
  float ss = 0.f;
  for (int i = threadIdx.x; i < HID; i += 256) { float v = x[base + i]; ss += v * v; }
#pragma unroll
  for (int off = 16; off >= 1; off >>= 1) ss += __shfl_xor(ss, off, 32);
  if ((threadIdx.x & 31) == 0) red[threadIdx.x >> 5] = ss;
  __syncthreads();
  float tot = 0.f;
#pragma unroll
  for (int i = 0; i < 8; ++i) tot += red[i];
  const float inv = rsqrtf(tot / (float)HID + EPSILON);
  for (int i = threadIdx.x; i < HID; i += 256)
    out[base + i] = resid[base + i] + x[base + i] * inv * (1.0f + wln[i]);
}

// ---------------------------------------------------------------- RoPE (fp32 in, bf16 out)
__global__ __launch_bounds__(256) void rope_kernel(const float* __restrict__ x,
                                                   const float* __restrict__ cosT,
                                                   const float* __restrict__ sinT,
                                                   bf16* __restrict__ out,
                                                   int nheads, float scale) {
  const int d  = threadIdx.x;                 // 0..255
  const int hh = blockIdx.x % nheads;
  const int rs = blockIdx.x / nheads;         // token row 0..MROWS-1
  const int s  = rs % SEQ;
  const size_t base = (size_t)rs * ((size_t)nheads * HD) + (size_t)hh * HD;
  const float v  = x[base + d];
  const float vr = (d < 128) ? -x[base + d + 128] : x[base + d - 128];
  const float c  = cosT[(size_t)s * HD + d];
  const float sn = sinT[(size_t)s * HD + d];
  out[base + d] = f2bf((v * c + vr * sn) * scale);
}

// ---------------------------------------------------------------- WMMA GEMM (NT): C[M,N] = A[M,K] * W[N,K]^T
// Block: 256 thr (8 waves). Tile 128(M) x 256(N), BK=32, double-buffered LDS
// staged with async global->LDS. Wave = 32x128 (2x8 WMMA tiles).
// mode 0: C fp32   mode 1: Cbf bf16   mode 2: Cbf = gelu(aux) * acc (bf16)
constexpr int LDT = 40;  // LDS row stride (elems): 80B, 16B-aligned, bank-spread

__device__ __forceinline__ void stage_tile(const bf16* __restrict__ A,
                                           const bf16* __restrict__ Bw,
                                           int K, int bm, int bn, int k0,
                                           bf16* as, bf16* bs, int tid) {
#pragma unroll
  for (int j = 0; j < 2; ++j) {                 // A: 128x32 = 512 x uint4
    const int idx = tid + j * 256;
    const int row = idx >> 2, cc = (idx & 3) * 8;
    cp16_g2l(A + (size_t)(bm + row) * K + k0 + cc, as + row * LDT + cc);
  }
#pragma unroll
  for (int j = 0; j < 4; ++j) {                 // B: 256x32 = 1024 x uint4
    const int idx = tid + j * 256;
    const int row = idx >> 2, cc = (idx & 3) * 8;
    cp16_g2l(Bw + (size_t)(bn + row) * K + k0 + cc, bs + row * LDT + cc);
  }
}

__global__ __launch_bounds__(256) void gemm_bf16_wmma(const bf16* __restrict__ A,
                                                      const bf16* __restrict__ Bw,
                                                      float* __restrict__ C,
                                                      bf16* __restrict__ Cbf,
                                                      const float* __restrict__ aux,
                                                      int M, int N, int K, int mode) {
  __shared__ bf16 As[2 * 128 * LDT];   // 20 KB
  __shared__ bf16 Bs[2 * 256 * LDT];   // 40 KB

  const int tid   = threadIdx.x;
  const int lane  = tid & 31;
  const int w     = tid >> 5;
  const int wr    = w >> 1;        // 0..3 : row group of 32
  const int wc    = w & 1;         // 0..1 : col group of 128
  const int lhalf = lane >> 4;     // 0/1
  const int l16   = lane & 15;
  const int bm    = blockIdx.y * 128;
  const int bn    = blockIdx.x * 256;

  const v8f vzero = {0.f, 0.f, 0.f, 0.f, 0.f, 0.f, 0.f, 0.f};
  v8f acc[2][8];
#pragma unroll
  for (int m = 0; m < 2; ++m)
#pragma unroll
    for (int n = 0; n < 8; ++n) acc[m][n] = vzero;

  const int nsteps = K >> 5;

  // prologue: stage first tile into buffer 0
  stage_tile(A, Bw, K, bm, bn, 0, As, Bs, tid);
  wait_async_lds();
  __syncthreads();

  for (int s = 0; s < nsteps; ++s) {
    const int buf = s & 1;
    bf16* as = &As[buf * (128 * LDT)];
    bf16* bs = &Bs[buf * (256 * LDT)];
    // kick off next tile into the other buffer (overlaps with WMMA below)
    if (s + 1 < nsteps)
      stage_tile(A, Bw, K, bm, bn, (s + 1) * 32,
                 &As[(buf ^ 1) * (128 * LDT)], &Bs[(buf ^ 1) * (256 * LDT)], tid);

    // A fragments: lane holds row M=l16; K = [kbA..kbA+7] U [16+kbA..16+kbA+7]
    const int kbA = lhalf * 8;
    v16bf aF[2];
#pragma unroll
    for (int m = 0; m < 2; ++m) {
      const int r = wr * 32 + m * 16 + l16;
      aF[m] = pack2(*(const uint4*)(&as[r * LDT + kbA]),
                    *(const uint4*)(&as[r * LDT + kbA + 16]));
    }
    // B fragments (lane holds col N=l16; K = [kbB..kbB+7] U [16+kbB..16+kbB+7]), consumed immediately
    const int kbB = lhalf * 8;
#pragma unroll
    for (int n = 0; n < 8; ++n) {
      const int cidx = wc * 128 + n * 16 + l16;
      const v16bf bF = pack2(*(const uint4*)(&bs[cidx * LDT + kbB]),
                             *(const uint4*)(&bs[cidx * LDT + kbB + 16]));
      acc[0][n] = wmma_bf16(aF[0], bF, acc[0][n]);
      acc[1][n] = wmma_bf16(aF[1], bF, acc[1][n]);
    }
    wait_async_lds();   // next tile resident before anyone flips buffers
    __syncthreads();
  }

  // epilogue: C frag layout — lane col = l16, VGPR i row = i + 8*lhalf
#pragma unroll
  for (int m = 0; m < 2; ++m)
#pragma unroll
    for (int n = 0; n < 8; ++n)
#pragma unroll
      for (int i = 0; i < 8; ++i) {
        const int row = bm + wr * 32 + m * 16 + i + 8 * lhalf;
        const int col = bn + wc * 128 + n * 16 + l16;
        const size_t off = (size_t)row * N + col;
        const float v = acc[m][n][i];
        if (mode == 0)      C[off]  = v;
        else if (mode == 1) Cbf[off] = f2bf(v);
        else                Cbf[off] = f2bf(gelu_tanh(aux[off]) * v);
      }
}

// ---------------------------------------------------------------- flash attention (sliding window, softcap)
// Block: 256 thr (8 waves) x 128 queries x 1 head. KV blocks of 32 keys.
constexpr int KS_LD = 264;  // K tile row stride (256 dims + pad)
constexpr int VT_LD = 40;   // V^T tile row stride (32 keys + pad)
constexpr int PS_LD = 40;   // P tile row stride

__global__ __launch_bounds__(256) void attn_kernel(const bf16* __restrict__ qb,
                                                   const bf16* __restrict__ kb,
                                                   const bf16* __restrict__ vb,
                                                   bf16* __restrict__ ob) {
  __shared__ bf16 Ks[32 * KS_LD];    // [key][dim]
  __shared__ bf16 Vts[256 * VT_LD];  // [dim][key]  (transposed)
  __shared__ bf16 Ps[128 * PS_LD];   // [query][key]

  const int qt = blockIdx.x & 15;          // SEQ/128 = 16
  const int h  = (blockIdx.x >> 4) & 15;
  const int b  = blockIdx.x >> 8;

  const int tid   = threadIdx.x;
  const int lane  = tid & 31;
  const int w     = tid >> 5;
  const int lhalf = lane >> 4;
  const int l16   = lane & 15;

  const int    q0   = qt * 128;
  const size_t qrow = (size_t)(b * SEQ + q0 + w * 16 + l16);  // lane's query row (M = l16)
  const int    kvh  = h >> 1;                                 // GQA: 2 q-heads per kv-head

  const v8f vzero = {0.f, 0.f, 0.f, 0.f, 0.f, 0.f, 0.f, 0.f};
  v8f o[16];
#pragma unroll
  for (int t = 0; t < 16; ++t) o[t] = vzero;
  float mrow[8], lrow[8];
#pragma unroll
  for (int v = 0; v < 8; ++v) { mrow[v] = -1e30f; lrow[v] = 0.f; }

  int lo = q0 - (WIN - 1);
  if (lo < 0) lo = 0;
  lo &= ~31;

  for (int j0 = lo; j0 < q0 + 128; j0 += 32) {
    __syncthreads();
    // stage K [32 x 256] (async -> LDS) and V^T [256 x 32] (manual transpose)
    for (int i = tid; i < 1024; i += 256) {           // 1024 uint4 per tile
      const int kk = i >> 5;
      const int cc = (i & 31) * 8;
      const size_t grow = (size_t)(b * SEQ + j0 + kk) * ((size_t)NKV * HD) + (size_t)kvh * HD + cc;
      cp16_g2l(kb + grow, &Ks[kk * KS_LD + cc]);
      union { uint4 u; bf16 e[8]; } tv;
      tv.u = *(const uint4*)(vb + grow);
#pragma unroll
      for (int e = 0; e < 8; ++e) Vts[(cc + e) * VT_LD + kk] = tv.e[e];
    }
    wait_async_lds();
    __syncthreads();

    // scores S[16 x 32] per wave : Q(16x256) . K^T
    v8f sc[2];
    sc[0] = vzero; sc[1] = vzero;
#pragma unroll
    for (int c = 0; c < 8; ++c) {                     // 256 dims / 32
      const bf16* qp = qb + qrow * ((size_t)NH * HD) + (size_t)h * HD + c * 32 + lhalf * 8;
      const v16bf aF = pack2(*(const uint4*)qp, *(const uint4*)(qp + 16));
#pragma unroll
      for (int nt = 0; nt < 2; ++nt) {
        const int key = nt * 16 + l16;
        const uint4 blo = *(const uint4*)(&Ks[key * KS_LD + c * 32 + lhalf * 8]);
        const uint4 bhi = *(const uint4*)(&Ks[key * KS_LD + c * 32 + lhalf * 8 + 16]);
        sc[nt] = wmma_bf16(aF, pack2(blo, bhi), sc[nt]);
      }
    }

    // softcap + mask + online softmax (row r = v + 8*lhalf, col = j0 + nt*16 + l16)
#pragma unroll
    for (int v = 0; v < 8; ++v) {
      const int ig = q0 + w * 16 + v + 8 * lhalf;     // global query index
      float mx = -1e30f;
#pragma unroll
      for (int nt = 0; nt < 2; ++nt) {
        float s = sc[nt][v];                          // scaling folded into Q
        s = CAP * tanhf(s * (1.0f / CAP));
        const int j = j0 + nt * 16 + l16;
        if (j > ig || j + WIN <= ig) s = -1e30f;
        sc[nt][v] = s;
        mx = fmaxf(mx, s);
      }
#pragma unroll
      for (int off = 1; off < 16; off <<= 1) mx = fmaxf(mx, __shfl_xor(mx, off, 32));
      const float mnew  = fmaxf(mrow[v], mx);
      const float scale = __expf(mrow[v] - mnew);
      float ps = 0.f;
#pragma unroll
      for (int nt = 0; nt < 2; ++nt) {
        const float s = sc[nt][v];
        const float p = (s <= -1e29f) ? 0.f : __expf(s - mnew);
        sc[nt][v] = p;
        ps += p;
      }
#pragma unroll
      for (int off = 1; off < 16; off <<= 1) ps += __shfl_xor(ps, off, 32);
      lrow[v] = lrow[v] * scale + ps;
      mrow[v] = mnew;
#pragma unroll
      for (int t = 0; t < 16; ++t) o[t][v] *= scale;
      // stash P (bf16) for layout conversion C-frag -> A-frag
#pragma unroll
      for (int nt = 0; nt < 2; ++nt)
        Ps[(w * 16 + v + 8 * lhalf) * PS_LD + nt * 16 + l16] = f2bf(sc[nt][v]);
    }

    // O += P(16x32) . V(32x256)  — same-wave DS ordering makes Ps safe to re-read
    {
      const int prow = w * 16 + l16;
      const uint4 plo = *(const uint4*)(&Ps[prow * PS_LD + lhalf * 8]);
      const uint4 phi = *(const uint4*)(&Ps[prow * PS_LD + 16 + lhalf * 8]);
      const v16bf pF = pack2(plo, phi);
#pragma unroll
      for (int t = 0; t < 16; ++t) {
        const int d = t * 16 + l16;
        const uint4 vlo = *(const uint4*)(&Vts[d * VT_LD + lhalf * 8]);
        const uint4 vhi = *(const uint4*)(&Vts[d * VT_LD + lhalf * 8 + 16]);
        o[t] = wmma_bf16(pF, pack2(vlo, vhi), o[t]);
      }
    }
  }

  // normalize + store (C frag: row = v + 8*lhalf, col = t*16 + l16)
#pragma unroll
  for (int t = 0; t < 16; ++t)
#pragma unroll
    for (int v = 0; v < 8; ++v) {
      const size_t row = (size_t)(b * SEQ + q0 + w * 16 + v + 8 * lhalf);
      ob[row * ((size_t)NH * HD) + (size_t)h * HD + t * 16 + l16] = f2bf(o[t][v] / lrow[v]);
    }
}

// ---------------------------------------------------------------- launch
static inline void* bump(uint8_t*& p, size_t bytes) {
  void* r = (void*)p;
  p += (bytes + 255) & ~(size_t)255;
  return r;
}

extern "C" void kernel_launch(void* const* d_in, const int* in_sizes, int n_in,
                              void* d_out, int out_size, void* d_ws, size_t ws_size,
                              hipStream_t stream) {
  (void)in_sizes; (void)n_in; (void)out_size; (void)ws_size;
  const float* hidden = (const float*)d_in[0];
  const float* cosT   = (const float*)d_in[1];
  const float* sinT   = (const float*)d_in[2];
  const float* Wq     = (const float*)d_in[3];
  const float* Wk     = (const float*)d_in[4];
  const float* Wv     = (const float*)d_in[5];
  const float* Wo     = (const float*)d_in[6];
  const float* Wg     = (const float*)d_in[7];
  const float* Wu     = (const float*)d_in[8];
  const float* Wd     = (const float*)d_in[9];
  const float* w_in   = (const float*)d_in[10];
  const float* w_pa   = (const float*)d_in[11];
  const float* w_pf   = (const float*)d_in[12];
  const float* w_ff   = (const float*)d_in[13];
  float* out = (float*)d_out;

  uint8_t* p = (uint8_t*)d_ws;
  const size_t QO = (size_t)NH * HD;   // 4096
  const size_t KO = (size_t)NKV * HD;  // 2048
  bf16* Wq_bf = (bf16*)bump(p, QO * HID * 2);
  bf16* Wk_bf = (bf16*)bump(p, KO * HID * 2);
  bf16* Wv_bf = (bf16*)bump(p, KO * HID * 2);
  bf16* Wo_bf = (bf16*)bump(p, (size_t)HID * QO * 2);
  bf16* Wg_bf = (bf16*)bump(p, (size_t)FF * HID * 2);
  bf16* Wu_bf = (bf16*)bump(p, (size_t)FF * HID * 2);
  bf16* Wd_bf = (bf16*)bump(p, (size_t)HID * FF * 2);
  bf16* x_bf  = (bf16*)bump(p, (size_t)MROWS * HID * 2);
  float* qf   = (float*)bump(p, (size_t)MROWS * QO * 4);
  float* kf   = (float*)bump(p, (size_t)MROWS * KO * 4);
  bf16* q_bf  = (bf16*)bump(p, (size_t)MROWS * QO * 2);
  bf16* k_bf  = (bf16*)bump(p, (size_t)MROWS * KO * 2);
  bf16* v_bf  = (bf16*)bump(p, (size_t)MROWS * KO * 2);
  bf16* a_bf  = (bf16*)bump(p, (size_t)MROWS * QO * 2);
  float* attnW = (float*)bump(p, (size_t)MROWS * HID * 4);
  float* h2    = (float*)bump(p, (size_t)MROWS * HID * 4);
  bf16* y_bf   = (bf16*)bump(p, (size_t)MROWS * HID * 2);
  float* gateF = (float*)bump(p, (size_t)MROWS * FF * 4);
  bf16* act_bf = (bf16*)bump(p, (size_t)MROWS * FF * 2);
  float* mlpF  = (float*)bump(p, (size_t)MROWS * HID * 4);

  // weights -> bf16
  cvt_f32_bf16<<<4096, 256, 0, stream>>>(Wq, Wq_bf, (long)QO * HID);
  cvt_f32_bf16<<<4096, 256, 0, stream>>>(Wk, Wk_bf, (long)KO * HID);
  cvt_f32_bf16<<<4096, 256, 0, stream>>>(Wv, Wv_bf, (long)KO * HID);
  cvt_f32_bf16<<<4096, 256, 0, stream>>>(Wo, Wo_bf, (long)HID * QO);
  cvt_f32_bf16<<<8192, 256, 0, stream>>>(Wg, Wg_bf, (long)FF * HID);
  cvt_f32_bf16<<<8192, 256, 0, stream>>>(Wu, Wu_bf, (long)FF * HID);
  cvt_f32_bf16<<<8192, 256, 0, stream>>>(Wd, Wd_bf, (long)HID * FF);

  // input LN
  rmsnorm_bf16_kernel<<<MROWS, 256, 0, stream>>>(hidden, w_in, x_bf);

  // QKV projections  (grid: N/256 x M/128)
  gemm_bf16_wmma<<<dim3(QO / 256, MROWS / 128), 256, 0, stream>>>(
      x_bf, Wq_bf, qf, nullptr, nullptr, MROWS, (int)QO, HID, 0);
  gemm_bf16_wmma<<<dim3(KO / 256, MROWS / 128), 256, 0, stream>>>(
      x_bf, Wk_bf, kf, nullptr, nullptr, MROWS, (int)KO, HID, 0);
  gemm_bf16_wmma<<<dim3(KO / 256, MROWS / 128), 256, 0, stream>>>(
      x_bf, Wv_bf, nullptr, v_bf, nullptr, MROWS, (int)KO, HID, 1);

  // RoPE (fold attention scaling into Q)
  rope_kernel<<<MROWS * NH, 256, 0, stream>>>(qf, cosT, sinT, q_bf, NH, SCALE_Q);
  rope_kernel<<<MROWS * NKV, 256, 0, stream>>>(kf, cosT, sinT, k_bf, NKV, 1.0f);

  // sliding-window flash attention
  attn_kernel<<<BB * NH * (SEQ / 128), 256, 0, stream>>>(q_bf, k_bf, v_bf, a_bf);

  // output projection + post-attn residual/LN   (3584/256 = 14)
  gemm_bf16_wmma<<<dim3(HID / 256, MROWS / 128), 256, 0, stream>>>(
      a_bf, Wo_bf, attnW, nullptr, nullptr, MROWS, HID, (int)QO, 0);
  add_rmsnorm_f32<<<MROWS, 256, 0, stream>>>(attnW, hidden, w_pa, h2);

  // MLP: y = rms(h2); act = gelu(y Wg^T) * (y Wu^T); mlp = act Wd^T
  rmsnorm_bf16_kernel<<<MROWS, 256, 0, stream>>>(h2, w_pf, y_bf);
  gemm_bf16_wmma<<<dim3(FF / 256, MROWS / 128), 256, 0, stream>>>(
      y_bf, Wg_bf, gateF, nullptr, nullptr, MROWS, FF, HID, 0);
  gemm_bf16_wmma<<<dim3(FF / 256, MROWS / 128), 256, 0, stream>>>(
      y_bf, Wu_bf, nullptr, act_bf, gateF, MROWS, FF, HID, 2);
  gemm_bf16_wmma<<<dim3(HID / 256, MROWS / 128), 256, 0, stream>>>(
      act_bf, Wd_bf, mlpF, nullptr, nullptr, MROWS, HID, FF, 0);

  // final residual/LN -> fp32 output
  add_rmsnorm_f32<<<MROWS, 256, 0, stream>>>(mlpF, h2, w_ff, out);
}
